// NaiveSHA_17111149707849
// MI455X (gfx1250) — matrix operand order repaired and checked by reading, not compiled, not run on hardware
//
#include <hip/hip_runtime.h>
#include <hip/hip_bf16.h>

// ---------------------------------------------------------------------------
// CDNA5 (gfx1250) single-head attention, bf16 WMMA pipeline.
// wave32; v_wmma_f32_16x16x32_bf16 for every GEMM; double-buffered LDS tiles
// filled with GLOBAL_LOAD_ASYNC_TO_LDS_B128 (ASYNCcnt) when available.
// ---------------------------------------------------------------------------

typedef __attribute__((ext_vector_type(16))) __bf16       v16bf;
typedef __attribute__((ext_vector_type(8)))  float        v8f;
typedef __attribute__((ext_vector_type(4)))  unsigned int u32x4;
typedef __attribute__((ext_vector_type(4)))  int          s32x4;

union FragAB { v16bf v; u32x4 q[2]; };

#if defined(__has_builtin)
#if __has_builtin(__builtin_amdgcn_global_load_async_to_lds_b128) && \
    __has_builtin(__builtin_amdgcn_s_wait_asynccnt)
#define ASYNC_LDS 1
#endif
#endif
#ifndef ASYNC_LDS
#define ASYNC_LDS 0
#endif

#if ASYNC_LDS
// Builtin signature: (v4i addrspace(1)* gsrc, v4i addrspace(3)* ldst, Ii, Ii).
// Generic LDS pointer = {shared aperture, offset}; low 32 bits are the LDS
// byte offset (ISA flat-aperture rule), so integer truncation gives the
// addrspace(3) address. Global generic pointers are numerically AS1.
#define AS1P(p) ((__attribute__((address_space(1))) s32x4*)(unsigned long long)(p))
#define AS3P(p) ((__attribute__((address_space(3))) s32x4*)(unsigned int)(unsigned long long)(p))
#define ASYNC_CP16(g, l) \
  __builtin_amdgcn_global_load_async_to_lds_b128(AS1P(g), AS3P(l), 0, 0)
#define ASYNC_WAIT(n) __builtin_amdgcn_s_wait_asynccnt(n)
#else
#define ASYNC_WAIT(n)
#endif

__device__ __forceinline__ unsigned short f32_to_bf16(float f) {
  union { float f; unsigned int u; } c; c.f = f;
  unsigned int u = c.u;
  unsigned int r = u + 0x7FFFu + ((u >> 16) & 1u);   // round-to-nearest-even
  return (unsigned short)(r >> 16);
}

// -------------------------------- convert ----------------------------------
__global__ void cvt_f32_bf16_kernel(const float* __restrict__ src,
                                    unsigned short* __restrict__ dst,
                                    long long n) {
  long long i = (long long)blockIdx.x * blockDim.x + threadIdx.x;
  long long stride = (long long)gridDim.x * blockDim.x;
  for (; i < n; i += stride) dst[i] = f32_to_bf16(src[i]);
}

// --------------------------------- GEMM ------------------------------------
// C = A @ B^T (+bias), A:[M,K] bf16 row-major, B:[N,K] bf16 row-major.
// Block tile 128(M) x 128(N), K-step 32, double-buffered LDS.
// 8 waves arranged 4(M) x 2(N); each wave owns 32x64 = 8 wmma accumulators,
// so each K-step is 8 WMMAs fed by 2 A-fragments + 4 B-fragments.
#define BM 128
#define BN 128
#define BK 32
#define LDSK (BK + 8)     // +16B pad per row: keeps 16B alignment, spreads banks

enum { EPI_BF16 = 0, EPI_BF16_TRANS = 1, EPI_F32_BIAS = 2, EPI_F32_CAUSAL = 3 };

__global__ __launch_bounds__(256) void gemm_bf16_wmma(
    const unsigned short* __restrict__ A, long long aBatch, int lda,
    const unsigned short* __restrict__ B, long long bBatch, int ldb,
    const float* __restrict__ bias,
    void* __restrict__ C, long long cBatch, int ldc,
    int M, int N, int K,
    int epi, float scale, int aCausal, int transRows) {
  (void)M; (void)N;

  __shared__ unsigned short As[2 * BM * LDSK];
  __shared__ unsigned short Bs[2 * BN * LDSK];

  const int tid    = threadIdx.x;
  const int lane   = tid & 31;
  const int wave   = tid >> 5;
  const int waveM  = wave >> 1;          // 0..3 -> 32 M-rows each
  const int waveN  = wave & 1;           // 0..1 -> 64 N-cols each
  const int laneN  = lane & 15;
  const int laneHi = lane >> 4;          // half-wave select
  const int blockRow = blockIdx.y * BM;
  const int blockCol = blockIdx.x * BN;
  const int z = blockIdx.z;

  const unsigned short* Ab = A + (long long)z * aBatch;
  const unsigned short* Bb = B + (long long)z * bBatch;

  // Causal score tile entirely above the diagonal: emit -inf, skip the GEMM.
  if (epi == EPI_F32_CAUSAL && blockCol >= blockRow + BM) {
    float* Cb = (float*)C + (long long)z * cBatch;
    for (int mi = 0; mi < 2; ++mi)
      for (int j = 0; j < 4; ++j) {
        const int gn = blockCol + waveN * 64 + j * 16 + laneN;
        for (int e = 0; e < 8; ++e) {
          const int gm = blockRow + waveM * 32 + mi * 16 + e + 8 * laneHi;
          Cb[(long long)gm * ldc + gn] = -INFINITY;
        }
      }
    return;
  }

  v8f acc[2][4];
  for (int mi = 0; mi < 2; ++mi)
    for (int j = 0; j < 4; ++j)
      for (int e = 0; e < 8; ++e) acc[mi][j][e] = 0.f;

  // A lower-triangular (softmax probs): keys beyond this query block are zero.
  int kend = K;
  if (aCausal) { const int kc = blockRow + BM; kend = kc < K ? kc : K; }

  // Cooperative staging: 2 threads per row, 16 bf16 (2 x b128) per thread.
  const int srow  = tid >> 1;            // 0..127
  const int skoff = (tid & 1) * 16;      // 0 / 16

  auto stage = [&](int buf, int kt) {
    const unsigned short* ga = &Ab[(long long)(blockRow + srow) * lda + kt + skoff];
    const unsigned short* gb = &Bb[(long long)(blockCol + srow) * ldb + kt + skoff];
    unsigned short* la = &As[buf * BM * LDSK + srow * LDSK + skoff];
    unsigned short* lb = &Bs[buf * BN * LDSK + srow * LDSK + skoff];
#if ASYNC_LDS
    ASYNC_CP16(ga,     la);        // 4 async ops / stage -> ASYNCcnt += 4
    ASYNC_CP16(ga + 8, la + 8);
    ASYNC_CP16(gb,     lb);
    ASYNC_CP16(gb + 8, lb + 8);
#else
    *(u32x4*)la       = *(const u32x4*)ga;
    *(u32x4*)(la + 8) = *(const u32x4*)(ga + 8);
    *(u32x4*)lb       = *(const u32x4*)gb;
    *(u32x4*)(lb + 8) = *(const u32x4*)(gb + 8);
#endif
  };

  stage(0, 0);
  int cur = 0;

  for (int kt = 0; kt < kend; kt += BK) {
    const bool hasNext = (kt + BK) < kend;
    if (hasNext) stage(cur ^ 1, kt + BK);      // prefetch next tile into other buffer
    if (hasNext) { ASYNC_WAIT(4); } else { ASYNC_WAIT(0); }
    __syncthreads();                            // current tile visible to all waves

    const unsigned short* as = &As[cur * BM * LDSK];
    const unsigned short* bs = &Bs[cur * BN * LDSK];

    // A fragments (16x32, ISA layout): lanes 0-15 row m hold K 0-7 & 16-23,
    // lanes 16-31 hold K 8-15 & 24-31.
    FragAB fa[2];
    const int aklo = laneHi * 8;
#pragma unroll
    for (int mi = 0; mi < 2; ++mi) {
      const int am = waveM * 32 + mi * 16 + laneN;
      fa[mi].q[0] = *(const u32x4*)&as[am * LDSK + aklo];
      fa[mi].q[1] = *(const u32x4*)&as[am * LDSK + aklo + 16];
    }

#pragma unroll
    for (int j = 0; j < 4; ++j) {
      // B fragment (32x16): lanes 0-15 col n hold K 0-7 & 16-23; lanes 16-31 K 8-15 & 24-31.
      FragAB fb;
      const int bn = waveN * 64 + j * 16 + laneN;
      const int bk = laneHi * 8;
      fb.q[0] = *(const u32x4*)&bs[bn * LDSK + bk];
      fb.q[1] = *(const u32x4*)&bs[bn * LDSK + bk + 16];
#pragma unroll
      for (int mi = 0; mi < 2; ++mi)
        acc[mi][j] = __builtin_amdgcn_wmma_f32_16x16x32_bf16(
            false, fa[mi].v, false, fb.v, (short)0, acc[mi][j], false, false);
    }
    __syncthreads();            // everyone done reading before buffer reuse
    cur ^= 1;
  }

  // ------------------------------ epilogue ---------------------------------
  for (int mi = 0; mi < 2; ++mi)
    for (int j = 0; j < 4; ++j) {
      const int gn = blockCol + waveN * 64 + j * 16 + laneN;
      const float badd = (bias != nullptr && epi != EPI_F32_CAUSAL) ? bias[gn] : 0.f;
      for (int e = 0; e < 8; ++e) {
        const int gm = blockRow + waveM * 32 + mi * 16 + e + 8 * laneHi;
        const float v = acc[mi][j][e] + badd;
        if (epi == EPI_BF16) {
          unsigned short* Cb = (unsigned short*)C + (long long)z * cBatch;
          Cb[(long long)gm * ldc + gn] = f32_to_bf16(v);
        } else if (epi == EPI_BF16_TRANS) {   // per-batch transposed store (V^T)
          const int bb = gm / transRows;
          const int s  = gm - bb * transRows;
          unsigned short* Cb = (unsigned short*)C;
          Cb[(long long)bb * cBatch + (long long)gn * ldc + s] = f32_to_bf16(v);
        } else if (epi == EPI_F32_BIAS) {
          float* Cb = (float*)C + (long long)z * cBatch;
          Cb[(long long)gm * ldc + gn] = v;
        } else {                              // scale * |s|, causal mask
          float* Cb = (float*)C + (long long)z * cBatch;
          const float m = fabsf(acc[mi][j][e]) * scale;
          Cb[(long long)gm * ldc + gn] = (gn > gm) ? -INFINITY : m;
        }
      }
    }
}

// ------------------------------- softmax -----------------------------------
// One 256-thread block per row of 2048 fp32 scores -> bf16 probabilities.
__global__ __launch_bounds__(256) void softmax_rows_kernel(
    const float* __restrict__ s, unsigned short* __restrict__ p, int n) {
  const long long row = blockIdx.x;
  const float* sr = s + row * (long long)n;
  unsigned short* pr = p + row * (long long)n;
  const int t = threadIdx.x, ln = t & 31, wv = t >> 5;

  float vals[8];
  float mx = -INFINITY;
#pragma unroll
  for (int e = 0; e < 8; ++e) { vals[e] = sr[t + e * 256]; mx = fmaxf(mx, vals[e]); }
  for (int off = 16; off; off >>= 1) mx = fmaxf(mx, __shfl_xor(mx, off, 32));

  __shared__ float redm[8];
  __shared__ float reds[8];
  if (ln == 0) redm[wv] = mx;
  __syncthreads();
  mx = redm[0];
#pragma unroll
  for (int w = 1; w < 8; ++w) mx = fmaxf(mx, redm[w]);

  float sum = 0.f;
#pragma unroll
  for (int e = 0; e < 8; ++e) { vals[e] = __expf(vals[e] - mx); sum += vals[e]; }
  for (int off = 16; off; off >>= 1) sum += __shfl_xor(sum, off, 32);
  if (ln == 0) reds[wv] = sum;
  __syncthreads();
  sum = 0.f;
#pragma unroll
  for (int w = 0; w < 8; ++w) sum += reds[w];

  const float inv = 1.f / sum;
#pragma unroll
  for (int e = 0; e < 8; ++e) pr[t + e * 256] = f32_to_bf16(vals[e] * inv);
}

// ------------------------------ launcher -----------------------------------
extern "C" void kernel_launch(void* const* d_in, const int* in_sizes, int n_in,
                              void* d_out, int out_size, void* d_ws, size_t ws_size,
                              hipStream_t stream) {
  (void)in_sizes; (void)n_in; (void)out_size; (void)ws_size;

  const float* x  = (const float*)d_in[0];
  const float* Wq = (const float*)d_in[1];
  const float* bq = (const float*)d_in[2];
  const float* Wk = (const float*)d_in[3];
  const float* bk = (const float*)d_in[4];
  const float* Wv = (const float*)d_in[5];
  const float* bv = (const float*)d_in[6];
  const float* Wo = (const float*)d_in[7];
  const float* bo = (const float*)d_in[8];
  float* out = (float*)d_out;

  const int Bz = 4, S = 2048, D = 1024, KD = 1024, VD = 1024;
  const int MS = Bz * S;                                   // 8192 rows

  char* ws = (char*)d_ws;
  size_t off = 0;
  auto take = [&](size_t bytes) -> char* {
    char* p = ws + off;
    off = (off + bytes + 255) & ~(size_t)255;
    return p;
  };
  unsigned short* xb  = (unsigned short*)take((size_t)MS * D * 2);
  unsigned short* wqb = (unsigned short*)take((size_t)KD * D * 2);
  unsigned short* wkb = (unsigned short*)take((size_t)KD * D * 2);
  unsigned short* wvb = (unsigned short*)take((size_t)VD * D * 2);
  unsigned short* wob = (unsigned short*)take((size_t)D * VD * 2);
  unsigned short* qb  = (unsigned short*)take((size_t)MS * KD * 2);
  unsigned short* kbf = (unsigned short*)take((size_t)MS * KD * 2);
  unsigned short* vt  = (unsigned short*)take((size_t)MS * VD * 2);   // [B][VD][S]
  float*          sc  = (float*)take((size_t)Bz * S * S * 4);
  unsigned short* pb  = (unsigned short*)take((size_t)Bz * S * S * 2);
  unsigned short* attn = xb;   // x is dead after the QKV projections: reuse

  const dim3 blk(256);

  // fp32 -> bf16
  cvt_f32_bf16_kernel<<<dim3(1024), blk, 0, stream>>>(x,  xb,  (long long)MS * D);
  cvt_f32_bf16_kernel<<<dim3(256),  blk, 0, stream>>>(Wq, wqb, (long long)KD * D);
  cvt_f32_bf16_kernel<<<dim3(256),  blk, 0, stream>>>(Wk, wkb, (long long)KD * D);
  cvt_f32_bf16_kernel<<<dim3(256),  blk, 0, stream>>>(Wv, wvb, (long long)VD * D);
  cvt_f32_bf16_kernel<<<dim3(256),  blk, 0, stream>>>(Wo, wob, (long long)D * VD);

  // Q = x@Wq^T + bq, K = x@Wk^T + bk  (bf16 out)
  gemm_bf16_wmma<<<dim3(KD / BN, MS / BM, 1), blk, 0, stream>>>(
      xb, 0, D, wqb, 0, D, bq, qb, 0, KD, MS, KD, D, EPI_BF16, 1.f, 0, 0);
  gemm_bf16_wmma<<<dim3(KD / BN, MS / BM, 1), blk, 0, stream>>>(
      xb, 0, D, wkb, 0, D, bk, kbf, 0, KD, MS, KD, D, EPI_BF16, 1.f, 0, 0);
  // V^T = (x@Wv^T + bv)^T per batch  -> vt[b][n][s]
  gemm_bf16_wmma<<<dim3(VD / BN, MS / BM, 1), blk, 0, stream>>>(
      xb, 0, D, wvb, 0, D, bv, vt, (long long)VD * S, S, MS, VD, D,
      EPI_BF16_TRANS, 1.f, 0, S);

  // scores_b = |Q_b @ K_b^T| / 32, causal -inf  (fp32 out, upper tiles skipped)
  gemm_bf16_wmma<<<dim3(S / BN, S / BM, Bz), blk, 0, stream>>>(
      qb, (long long)S * KD, KD, kbf, (long long)S * KD, KD, nullptr,
      sc, (long long)S * S, S, S, S, KD, EPI_F32_CAUSAL, 0.03125f, 0, 0);

  // row-wise softmax -> bf16 probabilities
  softmax_rows_kernel<<<dim3(Bz * S), blk, 0, stream>>>(sc, pb, S);

  // attn_b = P_b @ V_b  (= P_b @ (Vt_b)^T), P lower-triangular -> causal K bound
  gemm_bf16_wmma<<<dim3(VD / BN, S / BM, Bz), blk, 0, stream>>>(
      pb, (long long)S * S, S, vt, (long long)VD * S, S, nullptr,
      attn, (long long)S * VD, VD, S, VD, S, EPI_BF16, 1.f, 1, 0);

  // out = attn @ Wo^T + bo  (fp32 out)
  gemm_bf16_wmma<<<dim3(D / BN, MS / BM, 1), blk, 0, stream>>>(
      attn, 0, VD, wob, 0, VD, bo, out, 0, D, MS, D, VD, EPI_F32_BIAS, 1.f, 0, 0);
}
